// AVWGCN_64493228917304
// MI455X (gfx1250) — matrix operand order repaired and checked by reading, not compiled, not run on hardware
//
#include <hip/hip_runtime.h>
#include <hip/hip_bf16.h>

#define NN    2048
#define BB    32
#define CC    64
#define OO    64
#define DD    16
#define TWO_N 4096
#define KC    192   // K*C = 3*64

typedef __attribute__((ext_vector_type(16))) __bf16 v16bf;
typedef __attribute__((ext_vector_type(8)))  float  v8f;
typedef __attribute__((ext_vector_type(4)))  unsigned int uint4v;

union FragBF { v16bf v; uint4v u[2]; };

__device__ __forceinline__ unsigned short f2bf(float f) {
  union { float f; unsigned u; } x; x.f = f;
  unsigned r = x.u + 0x7FFFu + ((x.u >> 16) & 1u);
  return (unsigned short)(r >> 16);
}

// A operand: row-major [*, ld] bf16, tile row = `row` (already includes lane&15)
// layout: lanes 0-15 k = {k0..k0+7, k0+16..k0+23}; lanes 16-31 +8
__device__ __forceinline__ FragBF load_frag_a(const unsigned short* __restrict__ base,
                                              int row, int ld, int k0, int lane) {
  FragBF f;
  const unsigned short* p = base + (size_t)row * ld + k0 + ((lane >> 4) << 3);
  f.u[0] = *(const uint4v*)(p);
  f.u[1] = *(const uint4v*)(p + 16);
  return f;
}

// B operand from row-major B^T [*, ld] bf16, column = `col` (includes lane&15)
// layout: lanes 0-15 k = {k0..k0+7, k0+16..k0+23}; lanes 16-31 +8
__device__ __forceinline__ FragBF load_frag_bt(const unsigned short* __restrict__ base,
                                               int col, int ld, int k0, int lane) {
  FragBF f;
  const unsigned short* p = base + (size_t)col * ld + k0 + ((lane >> 4) << 3);
  f.u[0] = *(const uint4v*)(p);
  f.u[1] = *(const uint4v*)(p + 16);
  return f;
}

#define WMMA_BF16(A, B, Cacc) \
  __builtin_amdgcn_wmma_f32_16x16x32_bf16(false, (A), false, (B), (short)0, (Cacc), false, false)

// ---------------- conversion kernels ----------------
__global__ void k_conv_adj(const float* __restrict__ adj, unsigned short* __restrict__ Acat) {
  size_t i = (size_t)blockIdx.x * blockDim.x + threadIdx.x;
  if (i >= (size_t)NN * NN) return;
  size_t n = i >> 11, m = i & (NN - 1);
  Acat[n * TWO_N + NN + m] = f2bf(adj[i]);
}

__global__ void k_conv_sa(const float* __restrict__ sa, unsigned short* __restrict__ sab) {
  size_t i = (size_t)blockIdx.x * blockDim.x + threadIdx.x;
  if (i >= (size_t)NN * TWO_N) return;
  sab[i] = f2bf(sa[i]);
}

// x[b,n,c] -> xT[b,c,n] (B^T for x_g GEMM) and xg[n,b,0:64] (k=0 identity slot)
__global__ void k_conv_x(const float* __restrict__ x, unsigned short* __restrict__ xT,
                         unsigned short* __restrict__ xg) {
  size_t i = (size_t)blockIdx.x * blockDim.x + threadIdx.x;
  if (i >= (size_t)BB * NN * CC) return;
  int c = (int)(i & 63);
  int n = (int)((i >> 6) & (NN - 1));
  int b = (int)(i >> 17);
  unsigned short h = f2bf(x[i]);
  xT[((size_t)b * CC + c) * NN + n] = h;
  xg[((size_t)n * BB + b) * KC + c] = h;
}

// ---------------- supports = softmax(relu(E E^T)) per row ----------------
__global__ __launch_bounds__(256)
void k_supports(const float* __restrict__ E, float* __restrict__ supf,
                unsigned short* __restrict__ Acat) {
  __shared__ float e_s[DD];
  __shared__ float row[NN];
  __shared__ float red[256];
  int n = blockIdx.x, tid = threadIdx.x;
  if (tid < DD) e_s[tid] = E[n * DD + tid];
  __syncthreads();
  float lmax = 0.0f;  // relu output >= 0
  for (int m = tid; m < NN; m += 256) {
    const float* em = E + (size_t)m * DD;
    float d = 0.f;
#pragma unroll
    for (int t = 0; t < DD; ++t) d += e_s[t] * em[t];
    d = fmaxf(d, 0.f);
    row[m] = d;
    lmax = fmaxf(lmax, d);
  }
  red[tid] = lmax; __syncthreads();
  for (int s = 128; s > 0; s >>= 1) { if (tid < s) red[tid] = fmaxf(red[tid], red[tid + s]); __syncthreads(); }
  float rmax = red[0]; __syncthreads();
  float lsum = 0.f;
  for (int m = tid; m < NN; m += 256) { float v = __expf(row[m] - rmax); row[m] = v; lsum += v; }
  red[tid] = lsum; __syncthreads();
  for (int s = 128; s > 0; s >>= 1) { if (tid < s) red[tid] += red[tid + s]; __syncthreads(); }
  float inv = 1.0f / red[0];
  for (int m = tid; m < NN; m += 256) {
    float v = row[m] * inv;
    supf[(size_t)n * NN + m] = v;
    Acat[(size_t)n * TWO_N + m] = f2bf(v);
  }
}

// ---------------- GEMM 1: z = Acat @ sa^T, fused sigmoid blend ----------------
// writes S (bf16) and S^T (bf16)
__global__ __launch_bounds__(128)
void k_gemm_z(const unsigned short* __restrict__ Acat, const unsigned short* __restrict__ sab,
              const float* __restrict__ adj, const float* __restrict__ supf,
              unsigned short* __restrict__ S, unsigned short* __restrict__ ST) {
  int lane = threadIdx.x & 31, w = threadIdx.x >> 5;
  int m0 = blockIdx.x * 64;
  int n0 = blockIdx.y * 256 + w * 64;
  v8f zero = {0.f, 0.f, 0.f, 0.f, 0.f, 0.f, 0.f, 0.f};
  v8f acc[4][4];
#pragma unroll
  for (int i = 0; i < 4; ++i)
#pragma unroll
    for (int j = 0; j < 4; ++j) acc[i][j] = zero;
  int mr = lane & 15;
  for (int k0 = 0; k0 < TWO_N; k0 += 32) {
    FragBF a[4], b[4];
#pragma unroll
    for (int i = 0; i < 4; ++i) a[i] = load_frag_a(Acat, m0 + i * 16 + mr, TWO_N, k0, lane);
#pragma unroll
    for (int j = 0; j < 4; ++j) b[j] = load_frag_bt(sab, n0 + j * 16 + mr, TWO_N, k0, lane);
#pragma unroll
    for (int i = 0; i < 4; ++i)
#pragma unroll
      for (int j = 0; j < 4; ++j) acc[i][j] = WMMA_BF16(a[i].v, b[j].v, acc[i][j]);
  }
  int mh = (lane >> 4) * 8, nc = lane & 15;
#pragma unroll
  for (int i = 0; i < 4; ++i)
#pragma unroll
    for (int j = 0; j < 4; ++j)
#pragma unroll
      for (int r = 0; r < 8; ++r) {
        int m = m0 + i * 16 + mh + r;
        int n = n0 + j * 16 + nc;
        float zv = acc[i][j][r];
        float s2 = 1.f / (1.f + __expf(-zv));
        size_t idx = (size_t)m * NN + n;
        float v = s2 * adj[idx] + (1.f - s2) * supf[idx];
        unsigned short h = f2bf(v);
        S[idx] = h;
        ST[(size_t)n * NN + m] = h;
      }
}

// ---------------- GEMM 2: T2 = 2*S@S - I ----------------
__global__ __launch_bounds__(128)
void k_gemm_t2(const unsigned short* __restrict__ S, const unsigned short* __restrict__ ST,
               unsigned short* __restrict__ T2) {
  int lane = threadIdx.x & 31, w = threadIdx.x >> 5;
  int m0 = blockIdx.x * 64;
  int n0 = blockIdx.y * 256 + w * 64;
  v8f zero = {0.f, 0.f, 0.f, 0.f, 0.f, 0.f, 0.f, 0.f};
  v8f acc[4][4];
#pragma unroll
  for (int i = 0; i < 4; ++i)
#pragma unroll
    for (int j = 0; j < 4; ++j) acc[i][j] = zero;
  int mr = lane & 15;
  for (int k0 = 0; k0 < NN; k0 += 32) {
    FragBF a[4], b[4];
#pragma unroll
    for (int i = 0; i < 4; ++i) a[i] = load_frag_a(S, m0 + i * 16 + mr, NN, k0, lane);
#pragma unroll
    for (int j = 0; j < 4; ++j) b[j] = load_frag_bt(ST, n0 + j * 16 + mr, NN, k0, lane);
#pragma unroll
    for (int i = 0; i < 4; ++i)
#pragma unroll
      for (int j = 0; j < 4; ++j) acc[i][j] = WMMA_BF16(a[i].v, b[j].v, acc[i][j]);
  }
  int mh = (lane >> 4) * 8, nc = lane & 15;
#pragma unroll
  for (int i = 0; i < 4; ++i)
#pragma unroll
    for (int j = 0; j < 4; ++j)
#pragma unroll
      for (int r = 0; r < 8; ++r) {
        int m = m0 + i * 16 + mh + r;
        int n = n0 + j * 16 + nc;
        float v = 2.f * acc[i][j][r] - ((m == n) ? 1.f : 0.f);
        T2[(size_t)m * NN + n] = f2bf(v);
      }
}

// ---------------- GEMM 3: x_g[k] (k=1,2) = T_k @ x[b], into xg[n,b,k*64+c] ----------------
__global__ __launch_bounds__(128)
void k_gemm_xg(const unsigned short* __restrict__ S, const unsigned short* __restrict__ T2,
               const unsigned short* __restrict__ xT, unsigned short* __restrict__ xg) {
  int lane = threadIdx.x & 31, w = threadIdx.x >> 5;
  int p = blockIdx.z;
  int b = p >> 1;
  int kk = (p & 1) + 1;                 // 1 or 2
  const unsigned short* Ak = (kk == 1) ? S : T2;
  const unsigned short* BT = xT + (size_t)b * CC * NN;  // [c, n] rows
  int m0 = blockIdx.x * 256 + w * 64;   // node tile
  v8f zero = {0.f, 0.f, 0.f, 0.f, 0.f, 0.f, 0.f, 0.f};
  v8f acc[4][4];
#pragma unroll
  for (int i = 0; i < 4; ++i)
#pragma unroll
    for (int j = 0; j < 4; ++j) acc[i][j] = zero;
  int mr = lane & 15;
  for (int k0 = 0; k0 < NN; k0 += 32) {
    FragBF a[4], bfr[4];
#pragma unroll
    for (int i = 0; i < 4; ++i) a[i] = load_frag_a(Ak, m0 + i * 16 + mr, NN, k0, lane);
#pragma unroll
    for (int j = 0; j < 4; ++j) bfr[j] = load_frag_bt(BT, j * 16 + mr, NN, k0, lane);
#pragma unroll
    for (int i = 0; i < 4; ++i)
#pragma unroll
      for (int j = 0; j < 4; ++j) acc[i][j] = WMMA_BF16(a[i].v, bfr[j].v, acc[i][j]);
  }
  int mh = (lane >> 4) * 8, nc = lane & 15;
#pragma unroll
  for (int i = 0; i < 4; ++i)
#pragma unroll
    for (int j = 0; j < 4; ++j)
#pragma unroll
      for (int r = 0; r < 8; ++r) {
        int node = m0 + i * 16 + mh + r;
        int c = j * 16 + nc;
        xg[((size_t)node * BB + b) * KC + kk * 64 + c] = f2bf(acc[i][j][r]);
      }
}

// ---------------- final: per-node weight gen + [32x192]@[192x64] + bias ----------------
__global__ __launch_bounds__(128)
void k_final(const float* __restrict__ E, const float* __restrict__ pool,
             const float* __restrict__ bpool, const unsigned short* __restrict__ xg,
             float* __restrict__ out) {
  __shared__ float e_s[DD];
  __shared__ float W[KC * OO];                               // 48 KB fp32 [kc][o]
  __shared__ __align__(16) unsigned short WT[OO * KC];       // 24 KB bf16 [o][kc]
  __shared__ float bias_s[OO];
  int n = blockIdx.x, tid = threadIdx.x;
  if (tid < DD) e_s[tid] = E[n * DD + tid];
  __syncthreads();
  // W[kc][o] = sum_d e[d] * pool[d][kc][o]  (coalesced pool reads, L2-resident)
  for (int e = tid; e < KC * OO; e += 128) {
    float acc = 0.f;
#pragma unroll
    for (int d = 0; d < DD; ++d) acc += e_s[d] * pool[(size_t)d * (KC * OO) + e];
    W[e] = acc;
  }
  if (tid < OO) {
    float acc = 0.f;
#pragma unroll
    for (int d = 0; d < DD; ++d) acc += e_s[d] * bpool[d * OO + tid];
    bias_s[tid] = acc;
  }
  __syncthreads();
  for (int e = tid; e < KC * OO; e += 128) {
    int o = e & 63, kc = e >> 6;
    WT[o * KC + kc] = f2bf(W[e]);                            // B^T in LDS
  }
  __syncthreads();

  int lane = threadIdx.x & 31, w = threadIdx.x >> 5;
  const unsigned short* An = xg + (size_t)n * BB * KC;       // rows = batch, 192 bf16 each
  v8f acc0 = {0.f, 0.f, 0.f, 0.f, 0.f, 0.f, 0.f, 0.f};
  v8f acc1 = acc0;
  int o0 = w * 16;
  int mr = lane & 15;
  for (int k0 = 0; k0 < KC; k0 += 32) {
    FragBF a0 = load_frag_a(An, mr, KC, k0, lane);
    FragBF a1 = load_frag_a(An, 16 + mr, KC, k0, lane);
    FragBF bf = load_frag_bt(WT, o0 + mr, KC, k0, lane);     // LDS b128 reads
    acc0 = WMMA_BF16(a0.v, bf.v, acc0);
    acc1 = WMMA_BF16(a1.v, bf.v, acc1);
  }
  int mh = (lane >> 4) * 8, nc = lane & 15;
  int o = o0 + nc;
  float bv = bias_s[o];
#pragma unroll
  for (int r = 0; r < 8; ++r) {
    int b0 = mh + r;
    int b1 = 16 + mh + r;
    out[((size_t)b0 * NN + n) * OO + o] = acc0[r] + bv;
    out[((size_t)b1 * NN + n) * OO + o] = acc1[r] + bv;
  }
}

// ---------------- launch ----------------
extern "C" void kernel_launch(void* const* d_in, const int* in_sizes, int n_in,
                              void* d_out, int out_size, void* d_ws, size_t ws_size,
                              hipStream_t stream) {
  const float* x     = (const float*)d_in[0];  // [B,N,C]
  const float* E     = (const float*)d_in[1];  // [N,D]
  const float* adj   = (const float*)d_in[2];  // [N,N]
  const float* pool  = (const float*)d_in[3];  // [D,K,C,O]
  const float* bpool = (const float*)d_in[4];  // [D,O]
  const float* sa    = (const float*)d_in[5];  // [N,2N]
  float* out = (float*)d_out;
  char* ws = (char*)d_ws;

  // workspace layout (bytes)
  size_t off = 0;
  unsigned short* Acat = (unsigned short*)(ws + off); off += (size_t)NN * TWO_N * 2;  // 16 MB
  unsigned short* sab  = (unsigned short*)(ws + off); off += (size_t)NN * TWO_N * 2;  // 16 MB
  float*          supf = (float*)(ws + off);          off += (size_t)NN * NN * 4;     // 16 MB
  unsigned short* S    = (unsigned short*)(ws + off); off += (size_t)NN * NN * 2;     // 8 MB
  unsigned short* ST   = (unsigned short*)(ws + off); off += (size_t)NN * NN * 2;     // 8 MB
  unsigned short* T2   = (unsigned short*)(ws + off); off += (size_t)NN * NN * 2;     // 8 MB
  unsigned short* xT   = (unsigned short*)(ws + off); off += (size_t)BB * CC * NN * 2;// 8 MB
  unsigned short* xg   = (unsigned short*)(ws + off); off += (size_t)NN * BB * KC * 2;// 24 MB
  if (off > ws_size) return;  // workspace too small; nothing safe to do

  k_conv_adj<<<(NN * NN) / 256, 256, 0, stream>>>(adj, Acat);
  k_conv_sa<<<(NN * TWO_N) / 256, 256, 0, stream>>>(sa, sab);
  k_conv_x<<<(BB * NN * CC) / 256, 256, 0, stream>>>(x, xT, xg);
  k_supports<<<NN, 256, 0, stream>>>(E, supf, Acat);

  dim3 g2(NN / 64, NN / 256);
  k_gemm_z<<<g2, 128, 0, stream>>>(Acat, sab, adj, supf, S, ST);
  k_gemm_t2<<<g2, 128, 0, stream>>>(S, ST, T2);

  dim3 g4(NN / 256, 1, BB * 2);
  k_gemm_xg<<<g4, 128, 0, stream>>>(S, T2, xT, xg);

  k_final<<<NN, 128, 0, stream>>>(E, pool, bpool, xg, out);
}
